// CustomFullyConnectedLayerGoogleTopK_65618510348677
// MI455X (gfx1250) — matrix operand order repaired and checked by reading, not compiled, not run on hardware
//
#include <hip/hip_runtime.h>
#include <math.h>
#include <stdint.h>

// ---------------------------------------------------------------------------
// Problem constants (from reference)
// ---------------------------------------------------------------------------
#define IN_F    4096
#define OUT_F   4096
#define BATCH   8192
#define KCONST  3687.0f        // ceil(0.9 * 4096)

typedef float  v2f  __attribute__((ext_vector_type(2)));
typedef float  v8f  __attribute__((ext_vector_type(8)));
typedef __bf16 v4bf __attribute__((ext_vector_type(4)));
typedef __bf16 v8bf __attribute__((ext_vector_type(8)));
typedef __bf16 v16bf __attribute__((ext_vector_type(16)));

// ---------------------------------------------------------------------------
// Kernel 1: a_topk = clip(K * softmax(alpha), 0, 1)   (4096 elems, 1 block)
// ---------------------------------------------------------------------------
__global__ __launch_bounds__(1024)
void topk_softmax_kernel(const float* __restrict__ alpha,
                         float* __restrict__ a_topk) {
    __shared__ float red[32];
    const int tid  = threadIdx.x;
    const int lane = tid & 31;
    const int wv   = tid >> 5;

    float v[4];
    float lmax = -INFINITY;
#pragma unroll
    for (int i = 0; i < 4; ++i) {
        v[i] = alpha[tid + i * 1024];
        lmax = fmaxf(lmax, v[i]);
    }
#pragma unroll
    for (int o = 16; o > 0; o >>= 1)
        lmax = fmaxf(lmax, __shfl_xor(lmax, o, 32));
    if (lane == 0) red[wv] = lmax;
    __syncthreads();
    float gmax = red[lane];
#pragma unroll
    for (int o = 16; o > 0; o >>= 1)
        gmax = fmaxf(gmax, __shfl_xor(gmax, o, 32));

    float e[4], lsum = 0.0f;
#pragma unroll
    for (int i = 0; i < 4; ++i) {
        e[i] = __expf(v[i] - gmax);
        lsum += e[i];
    }
#pragma unroll
    for (int o = 16; o > 0; o >>= 1)
        lsum += __shfl_xor(lsum, o, 32);
    __syncthreads();
    if (lane == 0) red[wv] = lsum;
    __syncthreads();
    float gsum = red[lane];
#pragma unroll
    for (int o = 16; o > 0; o >>= 1)
        gsum += __shfl_xor(gsum, o, 32);

    const float scale = KCONST / gsum;
#pragma unroll
    for (int i = 0; i < 4; ++i) {
        float t = e[i] * scale;
        a_topk[tid + i * 1024] = fminf(fmaxf(t, 0.0f), 1.0f);
    }
}

// ---------------------------------------------------------------------------
// Kernel 2: split X (fp32) into bf16 hi + bf16 lo residual, same layout.
// ---------------------------------------------------------------------------
__global__ __launch_bounds__(256)
void split_x_kernel(const float* __restrict__ src,
                    __bf16* __restrict__ hi,
                    __bf16* __restrict__ lo) {
    const size_t base = ((size_t)blockIdx.x * 256 + threadIdx.x) * 4;
    float4 val = *reinterpret_cast<const float4*>(&src[base]);
    float xs[4] = {val.x, val.y, val.z, val.w};
    v4bf h, l;
#pragma unroll
    for (int j = 0; j < 4; ++j) {
        __bf16 hj = (__bf16)xs[j];
        h[j] = hj;
        l[j] = (__bf16)(xs[j] - (float)hj);
    }
    *reinterpret_cast<v4bf*>(&hi[base]) = h;
    *reinterpret_cast<v4bf*>(&lo[base]) = l;
}

// ---------------------------------------------------------------------------
// Kernel 3: build W[n][k] = V[(n-k)&4095][k] * a[(n-k)&4095] as bf16 hi/lo.
// Diagonal band through LDS; both global sides coalesced along k.
// ---------------------------------------------------------------------------
__global__ __launch_bounds__(256)
void build_w_kernel(const float* __restrict__ V,
                    const float* __restrict__ a_topk,
                    __bf16* __restrict__ Wh,
                    __bf16* __restrict__ Wl) {
    __shared__ float band[127][66];
    const int k0 = blockIdx.x * 64;
    const int n0 = blockIdx.y * 64;
    const int t  = threadIdx.x;

#pragma unroll
    for (int p = 0; p < 32; ++p) {
        int idx = t + p * 256;
        if (idx < 127 * 64) {
            int d = idx >> 6;
            int c = idx & 63;
            int iv = (n0 - k0 - 63 + d) & (OUT_F - 1);
            band[d][c] = V[(size_t)iv * IN_F + k0 + c] * a_topk[iv];
        }
    }
    __syncthreads();

#pragma unroll
    for (int p = 0; p < 16; ++p) {
        int idx = t + p * 256;
        int nn = idx >> 6;
        int kk = idx & 63;
        float val = band[nn - kk + 63][kk];
        __bf16 h = (__bf16)val;
        __bf16 l = (__bf16)(val - (float)h);
        size_t off = (size_t)(n0 + nn) * IN_F + k0 + kk;
        Wh[off] = h;
        Wl[off] = l;
    }
}

// ---------------------------------------------------------------------------
// Kernel 4: Y = X * W^T, bf16x3 split precision, V_WMMA_F32_16X16X32_BF16.
// Staging now uses GLOBAL_LOAD_ASYNC_TO_LDS_B128 (ASYNCcnt): memory -> LDS
// with no VGPR round-trip; each wave waits its own ASYNCcnt to 0 before the
// workgroup barrier that publishes the buffer.
// ---------------------------------------------------------------------------
#define BM   128
#define BN   128
#define KT   32
#define LSTR 40                       // LDS row stride in halves (16B aligned)
#define TILE_H (128 * LSTR)           // halves per tile array

__device__ __forceinline__ v16bf frag16(const __bf16* p, int step) {
    v8bf c0 = *reinterpret_cast<const v8bf*>(p);
    v8bf c1 = *reinterpret_cast<const v8bf*>(p + step);
    return __builtin_shufflevector(c0, c1, 0, 1, 2, 3, 4, 5, 6, 7,
                                           8, 9, 10, 11, 12, 13, 14, 15);
}

// Async 16B copy global -> LDS (GV mode, SADDR=NULL). lds_off is the byte
// offset within the workgroup LDS allocation (= low 32 bits of flat pointer;
// no static LDS in this kernel so dynamic shared starts at offset 0).
__device__ __forceinline__ void async_cp_b128(unsigned lds_off, const void* g) {
    asm volatile("global_load_async_to_lds_b128 %0, %1, off"
                 :: "v"(lds_off), "v"((unsigned long long)(uintptr_t)g)
                 : "memory");
}
__device__ __forceinline__ void wait_async0() {
    asm volatile("s_wait_asynccnt 0x0" ::: "memory");
}

__global__ __launch_bounds__(256)
void wmma_gemm_bf16x3_kernel(const __bf16* __restrict__ Xh,
                             const __bf16* __restrict__ Xl,
                             const __bf16* __restrict__ Wh,
                             const __bf16* __restrict__ Wl,
                             float* __restrict__ Y) {
    extern __shared__ __align__(16) __bf16 smem[];
    // layout: [buf(2)][array: Ah,Al,Bh,Bl][128*LSTR halves]

    const int tid  = threadIdx.x;
    const int lane = tid & 31;
    const int wv   = tid >> 5;            // 0..7
    const int lm   = lane & 15;
    const int lh   = lane >> 4;           // 0/1 half-wave
    const int waveM = wv >> 2;            // 0..1 -> +64*waveM rows
    const int waveN = wv & 3;             // 0..3 -> +32*waveN cols

    const int m0 = blockIdx.y * BM;
    const int n0 = blockIdx.x * BN;

    v8f acc[4][2] = {};

    // stage one 128x32 hi/lo tile pair for A and B into buffer `buf`
    // via async global->LDS copies (8 x b128 per thread, ASYNCcnt-tracked)
    auto stage = [&](int kt, int buf) {
        const int k0 = kt * KT;
        const unsigned bufOff = (unsigned)(buf * 4 * TILE_H) * 2u; // bytes
#pragma unroll
        for (int p = 0; p < 2; ++p) {
            int f  = tid + p * 256;       // v8bf unit index, 0..511
            int m  = f >> 2;              // 0..127
            int c8 = (f & 3) << 3;        // 0,8,16,24
            size_t ga = (size_t)(m0 + m) * IN_F + k0 + c8;
            size_t gb = (size_t)(n0 + m) * IN_F + k0 + c8;
            unsigned ldst = bufOff + (unsigned)(m * LSTR + c8) * 2u;
            async_cp_b128(ldst + 0u * TILE_H * 2u, &Xh[ga]);
            async_cp_b128(ldst + 1u * TILE_H * 2u, &Xl[ga]);
            async_cp_b128(ldst + 2u * TILE_H * 2u, &Wh[gb]);
            async_cp_b128(ldst + 3u * TILE_H * 2u, &Wl[gb]);
        }
    };

    const int NT = IN_F / KT;             // 128 k-tiles
    stage(0, 0);
    for (int kt = 0; kt < NT; ++kt) {
        wait_async0();                    // own async copies for stage(kt) done
        __syncthreads();                  // all waves' copies visible;
                                          // compute(kt-1) reads retired
        if (kt + 1 < NT) stage(kt + 1, (kt + 1) & 1);

        const int buf = kt & 1;
        __bf16* AHb = smem + buf * 4 * TILE_H;
        __bf16* ALb = AHb + TILE_H;
        __bf16* BHb = AHb + 2 * TILE_H;
        __bf16* BLb = AHb + 3 * TILE_H;

        // B fragments: lane = column n (lm); half-wave lh holds K chunks
        // {8*lh..+7} and {16+8*lh..+7}
        v16bf bh[2], bl[2];
#pragma unroll
        for (int tj = 0; tj < 2; ++tj) {
            int row = waveN * 32 + tj * 16 + lm;
            bh[tj] = frag16(&BHb[row * LSTR + 8 * lh], 16);
            bl[tj] = frag16(&BLb[row * LSTR + 8 * lh], 16);
        }
        // A fragments: lane = row m (lm); half-wave lh holds K chunks
        // {8*lh..+7} and {16+8*lh..+7}
#pragma unroll
        for (int ti = 0; ti < 4; ++ti) {
            int row = waveM * 64 + ti * 16 + lm;
            v16bf ah = frag16(&AHb[row * LSTR + 8 * lh], 16);
            v16bf al = frag16(&ALb[row * LSTR + 8 * lh], 16);
#pragma unroll
            for (int tj = 0; tj < 2; ++tj) {
                acc[ti][tj] = __builtin_amdgcn_wmma_f32_16x16x32_bf16(
                    false, ah, false, bh[tj], (short)0, acc[ti][tj], false, false);
                acc[ti][tj] = __builtin_amdgcn_wmma_f32_16x16x32_bf16(
                    false, ah, false, bl[tj], (short)0, acc[ti][tj], false, false);
                acc[ti][tj] = __builtin_amdgcn_wmma_f32_16x16x32_bf16(
                    false, al, false, bh[tj], (short)0, acc[ti][tj], false, false);
            }
        }
    }

    // store: C/D layout: VGPR v -> row (v + 8*lh), col lm
#pragma unroll
    for (int ti = 0; ti < 4; ++ti) {
#pragma unroll
        for (int tj = 0; tj < 2; ++tj) {
            int mb = m0 + waveM * 64 + ti * 16 + 8 * lh;
            int nb = n0 + waveN * 32 + tj * 16 + lm;
#pragma unroll
            for (int v = 0; v < 8; ++v)
                Y[(size_t)(mb + v) * OUT_F + nb] = acc[ti][tj][v];
        }
    }
}

// ---------------------------------------------------------------------------
// Launch
//   d_in[0]=x [8192*4096] f32, d_in[1]=V [4096*4096] f32, d_in[2]=alpha [4096]
//   d_out = y [8192*4096] f32
//   d_ws layout: a_topk f32[4096] (16KB) | Xh,Xl bf16[8192*4096] (64MB each)
//                | Wh,Wl bf16[4096*4096] (32MB each)   => ~192MB total
// ---------------------------------------------------------------------------
extern "C" void kernel_launch(void* const* d_in, const int* in_sizes, int n_in,
                              void* d_out, int out_size, void* d_ws, size_t ws_size,
                              hipStream_t stream) {
    const float* x     = (const float*)d_in[0];
    const float* V     = (const float*)d_in[1];
    const float* alpha = (const float*)d_in[2];
    float* y = (float*)d_out;

    char* ws = (char*)d_ws;
    float*  a_topk = (float*)ws;                              // 16 KB
    __bf16* Xh = (__bf16*)(ws + (16ull << 10));               // 64 MB
    __bf16* Xl = (__bf16*)(ws + (16ull << 10) + (64ull << 20));
    __bf16* Wh = (__bf16*)(ws + (16ull << 10) + (128ull << 20));
    __bf16* Wl = (__bf16*)(ws + (16ull << 10) + (160ull << 20));

    topk_softmax_kernel<<<1, 1024, 0, stream>>>(alpha, a_topk);
    split_x_kernel<<<(BATCH * IN_F) / (4 * 256), 256, 0, stream>>>(x, Xh, Xl);
    build_w_kernel<<<dim3(IN_F / 64, OUT_F / 64), 256, 0, stream>>>(V, a_topk, Wh, Wl);

    const size_t lds_bytes = 8ull * TILE_H * sizeof(__bf16);  // 80 KB
    wmma_gemm_bf16x3_kernel<<<dim3(OUT_F / BN, BATCH / BM), 256, lds_bytes, stream>>>(
        Xh, Xl, Wh, Wl, y);
}
